// GNNLSTM_63522566308234
// MI455X (gfx1250) — matrix operand order repaired and checked by reading, not compiled, not run on hardware
//
#include <hip/hip_runtime.h>
#include <hip/hip_bf16.h>
#include <math.h>

// ---------------- problem constants (from reference) ----------------
#define NNODES 3000
#define NEDGES 30000
#define SEQ    12
#define IN_F   128
#define HID    256
#define GATES  (4 * HID)          // 1024
#define OUT_F  12
#define NT     (NNODES * SEQ)     // 36000 rows for the node-time matmuls

typedef __attribute__((ext_vector_type(16))) _Float16 v16h;
typedef __attribute__((ext_vector_type(8)))  _Float16 v8h;
typedef __attribute__((ext_vector_type(4)))  _Float16 v4h;
typedef __attribute__((ext_vector_type(8)))  float    v8f;
typedef __attribute__((ext_vector_type(4)))  float    v4f;

// =====================================================================
// WMMA GEMM:  C[M,N] (+)= A[M,K] @ W[N,K]^T  (+ bias[N])
//   A and W are pre-converted f16 in workspace; f32 accumulate via
//   v_wmma_f32_16x16x32_f16. One wave computes 32(M) x 64(N):
//   2 A fragments x 4 B fragments -> 8 WMMAs per 12 b128 loads per k-step
//   (B reused 2x, A reused 4x).
// Fragment layouts used here (wave32; contiguous k, same order in A and B):
//   A 16x32: lane L -> M=L&15; K elems b..b+15, b = (L<16?0:16)
//            -> two 16B global_load_b128 (lands as one 8-VGPR operand).
//   B 32x16: lane L -> N=L&15; K elems (L<16?0:16)+0..15, contiguous in the
//            torch [N,K] weight row -> one 32B v16h load (2 x b128, no movs).
//   C/D 16x16: VGPR r -> row m0 + r + (L<16?0:8), col n0 + (L&15).
// Out-of-range rows/cols are CLAMPED for loads (valid memory, values
// discarded by the guarded store) so the hot loop never touches EXEC.
// =====================================================================
template <bool ACC, bool BIAS>
__global__ __launch_bounds__(256) void gemm_wmma_f16(
    const _Float16* __restrict__ A, int lda,
    const _Float16* __restrict__ W, int ldb,
    const float* __restrict__ bias,
    float* __restrict__ C, int ldc,
    int M, int N, int K)
{
    const int lane = threadIdx.x & 31;
    const int wave = threadIdx.x >> 5;
    const int m0 = (blockIdx.y * 8 + wave) * 32;
    const int n0 = blockIdx.x * 64;
    if (m0 >= M) return;                    // wave-uniform exit (EXEC stays full)

    const int mrow = lane & 15;
    const int hi   = lane >> 4;             // 0: lanes 0-15, 1: lanes 16-31

    // clamped A row pointers for the two M sub-tiles (no branch in hot loop)
    const _Float16* __restrict__ ap[2];
#pragma unroll
    for (int i = 0; i < 2; ++i) {
        int am = m0 + i * 16 + mrow;
        if (am >= M) am = M - 1;
        ap[i] = A + (size_t)am * lda + hi * 16;
    }

    const _Float16* __restrict__ bp[4];
#pragma unroll
    for (int j = 0; j < 4; ++j) {
        int bn = n0 + j * 16 + mrow;
        if (bn >= N) bn = N - 1;                               // clamp, no branch
        bp[j] = W + (size_t)bn * ldb + hi * 16;
    }

    v8f acc[2][4];
#pragma unroll
    for (int i = 0; i < 2; ++i) {
#pragma unroll
        for (int j = 0; j < 4; ++j) {
            if (ACC) {
                const int col = n0 + j * 16 + mrow;
#pragma unroll
                for (int r = 0; r < 8; ++r) {
                    const int row = m0 + i * 16 + hi * 8 + r;
                    acc[i][j][r] = (row < M && col < N) ? C[(size_t)row * ldc + col] : 0.0f;
                }
            } else {
#pragma unroll
                for (int r = 0; r < 8; ++r) acc[i][j][r] = 0.0f;
            }
        }
    }

    for (int k0 = 0; k0 < K; k0 += 32) {
        v16h a[2];
#pragma unroll
        for (int i = 0; i < 2; ++i) {
            const v8h a0 = *(const v8h*)(ap[i] + k0);        // K = b .. b+7
            const v8h a1 = *(const v8h*)(ap[i] + k0 + 8);    // K = b+8 .. b+15
            a[i] = __builtin_shufflevector(a0, a1,
                0, 1, 2, 3, 4, 5, 6, 7, 8, 9, 10, 11, 12, 13, 14, 15);
        }
#pragma unroll
        for (int j = 0; j < 4; ++j) {
            const v16h b = *(const v16h*)(bp[j] + k0);  // 16 contiguous K halfs
#pragma unroll
            for (int i = 0; i < 2; ++i) {
                acc[i][j] = __builtin_amdgcn_wmma_f32_16x16x32_f16(
                    false, a[i], false, b, (short)0, acc[i][j], false, false);
            }
        }
    }

    // ---- store (guarded; optional bias) ----
#pragma unroll
    for (int j = 0; j < 4; ++j) {
        const int col = n0 + j * 16 + mrow;
        if (col >= N) continue;
        const float bv = BIAS ? bias[col] : 0.0f;
#pragma unroll
        for (int i = 0; i < 2; ++i) {
#pragma unroll
            for (int r = 0; r < 8; ++r) {
                const int row = m0 + i * 16 + hi * 8 + r;
                if (row < M) C[(size_t)row * ldc + col] = acc[i][j][r] + bv;
            }
        }
    }
}

// ---------------- elementwise / graph kernels ----------------
__global__ void zero_f(float* __restrict__ p, int n) {
    int i = blockIdx.x * blockDim.x + threadIdx.x;
    if (i < n) p[i] = 0.0f;
}

// f32 -> f16, 4 elements per thread (all our sizes are multiples of 4)
__global__ void cvt_f16(const float* __restrict__ s, _Float16* __restrict__ d, int n4) {
    int i = blockIdx.x * blockDim.x + threadIdx.x;
    if (i >= n4) return;
    const v4f f = ((const v4f*)s)[i];
    v4h h;
#pragma unroll
    for (int c = 0; c < 4; ++c) h[c] = (_Float16)f[c];
    ((v4h*)d)[i] = h;
}

__global__ void deg_scatter(const long long* __restrict__ ei, float* __restrict__ deg, int E) {
    int e = blockIdx.x * blockDim.x + threadIdx.x;
    if (e < E) atomicAdd(&deg[(int)ei[E + e]], 1.0f);   // dst row of edge_index
}

__global__ void dinv_k(const float* __restrict__ deg, float* __restrict__ dinv, int n) {
    int i = blockIdx.x * blockDim.x + threadIdx.x;
    if (i < n) dinv[i] = rsqrtf(deg[i] + 1.0f);         // +1 self-loop; always > 0
}

// out[n,t,h] = bias[h] + dinv[n]^2 * lin[n,t,h]   (self-loop term + bias init)
__global__ void gcn_self_init(const float* __restrict__ lin, const float* __restrict__ dinv,
                              const float* __restrict__ bias, float* __restrict__ out) {
    int i = blockIdx.x * blockDim.x + threadIdx.x;
    if (i >= NT * HID) return;
    const int h = i & (HID - 1);
    const int n = i / (SEQ * HID);
    const float d = dinv[n];
    out[i] = bias[h] + d * d * lin[i];
}

// out[dst] += dinv[src]*dinv[dst] * lin[src]   over all [SEQ,HID] features
__global__ void gcn_scatter(const float* __restrict__ lin, const long long* __restrict__ ei,
                            const float* __restrict__ dinv, float* __restrict__ out) {
    const int e = blockIdx.x;
    const int s = (int)ei[e];
    const int d = (int)ei[NEDGES + e];
    const float nrm = dinv[s] * dinv[d];
    const v4f* __restrict__ sp = (const v4f*)(lin + (size_t)s * (SEQ * HID));
    float* __restrict__ dp = out + (size_t)d * (SEQ * HID);
    for (int i = threadIdx.x; i < (SEQ * HID) / 4; i += blockDim.x) {
        const v4f v = sp[i];
        atomicAdd(&dp[4 * i + 0], nrm * v[0]);
        atomicAdd(&dp[4 * i + 1], nrm * v[1]);
        atomicAdd(&dp[4 * i + 2], nrm * v[2]);
        atomicAdd(&dp[4 * i + 3], nrm * v[3]);
    }
}

__global__ void relu_k(float* __restrict__ p, int n) {
    int i = blockIdx.x * blockDim.x + threadIdx.x;
    if (i < n) p[i] = fmaxf(p[i], 0.0f);
}

__device__ __forceinline__ float sigm(float x) { return 1.0f / (1.0f + __expf(-x)); }

// gates hold x@W_ih^T + b_ih + h@W_hh^T ; add b_hh, update (h,c)
__global__ void lstm_cell(const float* __restrict__ gates, const float* __restrict__ b_hh,
                          float* __restrict__ hst, float* __restrict__ cst) {
    int idx = blockIdx.x * blockDim.x + threadIdx.x;
    if (idx >= NNODES * HID) return;
    const int j = idx & (HID - 1);
    const int n = idx / HID;
    const float* g = gates + (size_t)n * GATES;
    const float gi = sigm (g[j]           + b_hh[j]);
    const float gf = sigm (g[HID + j]     + b_hh[HID + j]);
    const float gg = tanhf(g[2 * HID + j] + b_hh[2 * HID + j]);
    const float go = sigm (g[3 * HID + j] + b_hh[3 * HID + j]);
    const float c  = gf * cst[idx] + gi * gg;
    cst[idx] = c;
    hst[idx] = go * tanhf(c);
}

// ---------------- host-side helpers ----------------
template <bool ACC, bool BIAS>
static inline void launch_gemm(const _Float16* A, int lda, const _Float16* W, int ldb,
                               const float* bias, float* C, int ldc,
                               int M, int N, int K, hipStream_t s) {
    dim3 grid((N + 63) / 64, (M + 255) / 256);   // 8 waves x 32 rows per block
    gemm_wmma_f16<ACC, BIAS><<<grid, 256, 0, s>>>(A, lda, W, ldb, bias, C, ldc, M, N, K);
}

static inline int cdiv(int a, int b) { return (a + b - 1) / b; }

static inline void cvt(const float* s, _Float16* d, int n, hipStream_t st) {
    cvt_f16<<<cdiv(n / 4, 256), 256, 0, st>>>(s, d, n / 4);
}

extern "C" void kernel_launch(void* const* d_in, const int* in_sizes, int n_in,
                              void* d_out, int out_size, void* d_ws, size_t ws_size,
                              hipStream_t stream) {
    (void)in_sizes; (void)n_in; (void)out_size; (void)ws_size;

    const float*     x    = (const float*)d_in[0];           // [3000,12,128]
    const long long* ei   = (const long long*)d_in[1];       // [2,30000] int64
    const float*     W1   = (const float*)d_in[2];           // [256,128]
    const float*     b1   = (const float*)d_in[3];
    const float*     W2   = (const float*)d_in[4];           // [256,256]
    const float*     b2   = (const float*)d_in[5];
    const float*     W_ih = (const float*)d_in[6];           // [1024,256]
    const float*     W_hh = (const float*)d_in[7];           // [1024,256]
    const float*     b_ih = (const float*)d_in[8];
    const float*     b_hh = (const float*)d_in[9];
    const float*     W_fc = (const float*)d_in[10];          // [12,256]
    const float*     b_fc = (const float*)d_in[11];
    float*           out  = (float*)d_out;                   // [3000,12]

    // ---- workspace layout (float offsets, all 64B aligned); ~95 MB ----
    float* ws = (float*)d_ws;
    float*    lin   = ws;                            // [36000,256] f32 GEMM out
    float*    gates = ws;                            // [3000,1024] (lin dead)
    float*    hst   = ws + 4608000;                  // [3000,256]
    float*    cst   = ws + 5376000;                  // [3000,256]
    float*    agg   = ws + 9216000;                  // [36000,256] f32 (h1, then h2)
    _Float16* ha    = (_Float16*)(ws + 18432000);    // 9,216,000 halfs: x/h1/h2 f16
    _Float16* hh    = (_Float16*)(ws + 23040000);    // 768,000 halfs: h state f16
    _Float16* w1h   = (_Float16*)(ws + 23424000);    // 32,768 halfs
    _Float16* w2h   = (_Float16*)(ws + 23440384);    // 65,536 halfs
    _Float16* wihh  = (_Float16*)(ws + 23473152);    // 262,144 halfs
    _Float16* whhh  = (_Float16*)(ws + 23604224);    // 262,144 halfs
    _Float16* wfch  = (_Float16*)(ws + 23735296);    // 3,072 halfs
    float*    deg   = ws + 23736832;                 // [3000]
    float*    dinv  = deg + NNODES;                  // [3000]

    // ---- GCN normalization: deg (with self-loop) -> dinv ----
    zero_f<<<cdiv(NNODES, 256), 256, 0, stream>>>(deg, NNODES);
    deg_scatter<<<cdiv(NEDGES, 256), 256, 0, stream>>>(ei, deg, NEDGES);
    dinv_k<<<cdiv(NNODES, 256), 256, 0, stream>>>(deg, dinv, NNODES);

    // ---- one-time f16 conversion of weights ----
    cvt(W1,   w1h,  HID * IN_F,   stream);
    cvt(W2,   w2h,  HID * HID,    stream);
    cvt(W_ih, wihh, GATES * HID,  stream);
    cvt(W_hh, whhh, GATES * HID,  stream);
    cvt(W_fc, wfch, OUT_F * HID,  stream);

    // ---- GCN layer 1: lin = x @ W1^T ; agg = relu(aggregate(lin) + b1) ----
    cvt(x, ha, NT * IN_F, stream);
    launch_gemm<false, false>(ha, IN_F, w1h, IN_F, nullptr, lin, HID, NT, HID, IN_F, stream);
    gcn_self_init<<<cdiv(NT * HID, 256), 256, 0, stream>>>(lin, dinv, b1, agg);
    gcn_scatter<<<NEDGES, 256, 0, stream>>>(lin, ei, dinv, agg);
    relu_k<<<cdiv(NT * HID, 256), 256, 0, stream>>>(agg, NT * HID);

    // ---- GCN layer 2 (agg f32 -> ha f16, then reuse agg as layer-2 target) ----
    cvt(agg, ha, NT * HID, stream);
    launch_gemm<false, false>(ha, HID, w2h, HID, nullptr, lin, HID, NT, HID, HID, stream);
    gcn_self_init<<<cdiv(NT * HID, 256), 256, 0, stream>>>(lin, dinv, b2, agg);
    gcn_scatter<<<NEDGES, 256, 0, stream>>>(lin, ei, dinv, agg);
    relu_k<<<cdiv(NT * HID, 256), 256, 0, stream>>>(agg, NT * HID);

    // ---- LSTM input sequence to f16; init states ----
    cvt(agg, ha, NT * HID, stream);
    zero_f<<<cdiv(NNODES * HID, 256), 256, 0, stream>>>(hst, NNODES * HID);
    zero_f<<<cdiv(NNODES * HID, 256), 256, 0, stream>>>(cst, NNODES * HID);

    for (int t = 0; t < SEQ; ++t) {
        cvt(hst, hh, NNODES * HID, stream);  // h_{t-1} -> f16
        // gates = h2[:, t, :] @ W_ih^T + b_ih   (A rows strided by SEQ*HID)
        launch_gemm<false, true>(ha + (size_t)t * HID, SEQ * HID, wihh, HID, b_ih,
                                 gates, GATES, NNODES, GATES, HID, stream);
        // gates += h @ W_hh^T
        launch_gemm<true, false>(hh, HID, whhh, HID, nullptr,
                                 gates, GATES, NNODES, GATES, HID, stream);
        lstm_cell<<<cdiv(NNODES * HID, 256), 256, 0, stream>>>(gates, b_hh, hst, cst);
    }

    // ---- FC head: out = h_T @ W_fc^T + b_fc   (N=12 handled by clamps/guards) ----
    cvt(hst, hh, NNODES * HID, stream);
    launch_gemm<false, true>(hh, HID, wfch, HID, b_fc, out, OUT_F,
                             NNODES, OUT_F, HID, stream);
}
